// CPAMDec_60361470378276
// MI455X (gfx1250) — matrix operand order repaired and checked by reading, not compiled, not run on hardware
//
#include <hip/hip_runtime.h>
#include <cstdint>

// ---------------------------------------------------------------------------
// Types for CDNA5 WMMA (wave32): v_wmma_f32_16x16x32_bf16
// ---------------------------------------------------------------------------
typedef __attribute__((ext_vector_type(8)))  __bf16 bf16x8;
typedef __attribute__((ext_vector_type(16))) __bf16 bf16x16;
typedef __attribute__((ext_vector_type(8)))  float  f32x8;

__device__ __forceinline__ bf16x16 join8(bf16x8 lo, bf16x8 hi) {
    union { bf16x16 v; bf16x8 p[2]; } u;
    u.p[0] = lo; u.p[1] = hi;
    return u.v;
}

// ---------------------------------------------------------------------------
// General GEMM: D[M,N] = epi(A[M,K] * B[K,N])
//   A row-major (lda), B row-major (ldb), D row-major (ldd)
//   epi: optional per-row scale/bias (folded BN), optional relu,
//        optional residual add (same layout as D)
//   blockIdx.z batches via strideA/strideB/strideD/strideR (0 = shared);
//   batch offsets are uniform 64-bit pre-adds to the base pointers, all
//   per-thread addressing uses 32-bit element offsets (saddr+voffset form;
//   every per-batch tensor here is < 2^31 elements).
//
// Bounds policy (branch-free):
//   - A rows >= M and B cols >= N are *clamped* (garbage flows only into
//     D rows/cols that the epilogue never writes).
//   - K tail must truly zero-pad; only the EDGEK=true instantiation pays for
//     it (clamped offset + cndmask select, still no exec-mask branches).
//   - EDGEK=false additionally requires lda % 4 == 0 (float4 A loads).
//
// Tile: 128x128 per block (256 thr = 8 wave32), 32x64 per wave, BK=32.
// Double-buffered LDS (ping-pong, one barrier per K-step): iteration i reads
// fragments from buf, stages tile i+1 into buf^1 (global latency + cvt VALU
// overlap the WMMA issue), then flips.
// A/B staged in LDS as bf16; B stored K-contiguous per column (transposed)
// so both fragment types load via 16B ds_load_b128 matching ISA VGPR layouts.
// ---------------------------------------------------------------------------
#define GEMM_BM 128
#define GEMM_BN 128
#define GEMM_BK 32
#define LDST    40   // padded LDS row stride (elements); 80B rows, 16B aligned

template<bool EDGEK>
__global__ __launch_bounds__(256) void gemm_bn_wmma(
    const float* __restrict__ A, long lda, long strideA,
    const float* __restrict__ B, long ldb, long strideB,
    float*       __restrict__ D, long ldd, long strideD,
    const float* __restrict__ scale,   // per-row, may be null
    const float* __restrict__ bias,    // per-row, may be null
    const float* __restrict__ resid, long strideR,  // may be null
    int M, int N, int K, int relu)
{
    const int bz = blockIdx.z;
    A += (long)bz * strideA;           // uniform 64-bit pre-adds
    B += (long)bz * strideB;
    D += (long)bz * strideD;
    if (resid) resid += (long)bz * strideR;

    const int ldai = (int)lda;
    const int ldbi = (int)ldb;
    const int lddi = (int)ldd;

    const int m0 = blockIdx.y * GEMM_BM;
    const int n0 = blockIdx.x * GEMM_BN;

    __shared__ __align__(16) __bf16 As[2][GEMM_BM][LDST];  // [m][k]
    __shared__ __align__(16) __bf16 Bs[2][GEMM_BN][LDST];  // [n][k] (B^T)

    const int tid  = threadIdx.x;
    const int wave = tid >> 5;
    const int lane = tid & 31;
    const int wm   = wave & 3;   // 0..3 -> M offset wm*32
    const int wn   = wave >> 2;  // 0..1 -> N offset wn*64
    const int h    = lane >> 4;  // half-wave select
    const int l16  = lane & 15;

    f32x8 acc[2][4];
#pragma unroll
    for (int i = 0; i < 2; ++i)
#pragma unroll
        for (int j = 0; j < 4; ++j)
            acc[i][j] = (f32x8){};

    // A-tile loader: thread -> (row = tid/2, 16 cols at (tid&1)*16)
    const int ar = tid >> 1;
    const int ac = (tid & 1) * 16;
    const int agr  = min(m0 + ar, M - 1);   // row clamp (see policy)
    const int aRow = agr * ldai;            // 32-bit element offset of row
    // B-tile loader: thread -> (col n = tid&127, 16 k's at (tid>>7)*16)
    const int bn_ = tid & 127;
    const int bk0 = (tid >> 7) * 16;
    const int bgn = min(n0 + bn_, N - 1);   // col clamp (see policy)

    // walking 32-bit offsets (fast path): one 32-bit add per staged tile
    const float4* __restrict__ A4 = (const float4*)A;
    int a4   = (aRow + ac) >> 2;            // float4 index, advance BK/4
    int bOff = bk0 * ldbi + bgn;            // element index, advance BK*ldb

    auto stage = [&](int buf, int k0) {
        // ---- A[agr, k0+ac .. +15] -> As (f32 -> bf16, branch-free) ----
        union { bf16x8 v[2]; __bf16 e[16]; } au;
        if (!EDGEK) {
#pragma unroll
            for (int q = 0; q < 4; ++q) {
                float4 f = A4[a4 + q];
                au.e[4 * q + 0] = (__bf16)f.x;
                au.e[4 * q + 1] = (__bf16)f.y;
                au.e[4 * q + 2] = (__bf16)f.z;
                au.e[4 * q + 3] = (__bf16)f.w;
            }
        } else {
#pragma unroll
            for (int e = 0; e < 16; ++e) {
                const int gk = k0 + ac + e;
                float v = A[aRow + min(gk, K - 1)];
                if (gk >= K) v = 0.0f;            // cndmask, no branch
                au.e[e] = (__bf16)v;
            }
        }
        *(bf16x8*)&As[buf][ar][ac]     = au.v[0]; // two ds_store_b128
        *(bf16x8*)&As[buf][ar][ac + 8] = au.v[1];

        // ---- B[k0+bk0.., bgn] -> Bs[n][k] (transposed, branch-free) ----
        union { bf16x8 v[2]; __bf16 e[16]; } bu;
#pragma unroll
        for (int e = 0; e < 16; ++e) {
            float v;
            if (!EDGEK) {
                v = B[bOff + e * ldbi];
            } else {
                const int gk = k0 + bk0 + e;
                v = B[min(gk, K - 1) * ldbi + bgn];
                if (gk >= K) v = 0.0f;
            }
            bu.e[e] = (__bf16)v;
        }
        *(bf16x8*)&Bs[buf][bn_][bk0]     = bu.v[0];
        *(bf16x8*)&Bs[buf][bn_][bk0 + 8] = bu.v[1];

        // advance walking offsets; prefetch one more tile ahead
        a4   += GEMM_BK / 4;
        bOff += GEMM_BK * ldbi;
        if (!EDGEK && k0 + GEMM_BK < K) {
            __builtin_prefetch(&A4[a4], 0, 1);        // global_prefetch_b8
            __builtin_prefetch(&B[bOff], 0, 1);
        }
    };

    // prologue: stage tile 0
    stage(0, 0);
    __syncthreads();

    int cur = 0;
    for (int k0 = GEMM_BK; k0 < K; k0 += GEMM_BK) {
        // ---- fragments from current buffer ----
        bf16x16 bfrag[4];
#pragma unroll
        for (int tn = 0; tn < 4; ++tn) {
            const int col = wn * 64 + tn * 16 + l16;
            const __bf16* bp = &Bs[cur][col][h * 8];
            bfrag[tn] = join8(*(const bf16x8*)bp, *(const bf16x8*)(bp + 16));
        }
        bf16x16 afrag[2];
#pragma unroll
        for (int tm = 0; tm < 2; ++tm) {
            const int rowl = wm * 32 + tm * 16 + l16;
            afrag[tm] = join8(*(const bf16x8*)&As[cur][rowl][h * 8],
                              *(const bf16x8*)&As[cur][rowl][16 + h * 8]);
        }
        // ---- stage next tile into the other buffer (overlaps WMMA) ----
        stage(cur ^ 1, k0);
        // ---- 8 WMMAs ----
#pragma unroll
        for (int tm = 0; tm < 2; ++tm)
#pragma unroll
            for (int tn = 0; tn < 4; ++tn)
                acc[tm][tn] = __builtin_amdgcn_wmma_f32_16x16x32_bf16(
                    false, afrag[tm], false, bfrag[tn],
                    (short)0, acc[tm][tn], false, false);
        __syncthreads();
        cur ^= 1;
    }

    // drain tile: fragments + WMMA, no further staging
    {
        bf16x16 bfrag[4];
#pragma unroll
        for (int tn = 0; tn < 4; ++tn) {
            const int col = wn * 64 + tn * 16 + l16;
            const __bf16* bp = &Bs[cur][col][h * 8];
            bfrag[tn] = join8(*(const bf16x8*)bp, *(const bf16x8*)(bp + 16));
        }
#pragma unroll
        for (int tm = 0; tm < 2; ++tm) {
            const int rowl = wm * 32 + tm * 16 + l16;
            bf16x16 afrag = join8(*(const bf16x8*)&As[cur][rowl][h * 8],
                                  *(const bf16x8*)&As[cur][rowl][16 + h * 8]);
#pragma unroll
            for (int tn = 0; tn < 4; ++tn)
                acc[tm][tn] = __builtin_amdgcn_wmma_f32_16x16x32_bf16(
                    false, afrag, false, bfrag[tn],
                    (short)0, acc[tm][tn], false, false);
        }
    }

    // ---- store: C/D layout: lane(l16,h), VGPR r -> row = h*8+r, col = l16
#pragma unroll
    for (int tm = 0; tm < 2; ++tm) {
#pragma unroll
        for (int tn = 0; tn < 4; ++tn) {
            const int col = n0 + wn * 64 + tn * 16 + l16;
            if (col >= N) continue;
            const int rbase = m0 + wm * 32 + tm * 16 + h * 8;
            int dOff = rbase * lddi + col;        // 32-bit walking offset
#pragma unroll
            for (int r = 0; r < 8; ++r) {
                const int row = rbase + r;
                if (row < M) {
                    float v = acc[tm][tn][r];
                    if (scale) v = v * scale[row] + bias[row];
                    if (resid) v += resid[dOff];
                    if (relu)  v = fmaxf(v, 0.0f);
                    D[dOff] = v;
                }
                dOff += lddi;
            }
        }
    }
}

// ---------------------------------------------------------------------------
// BN fold: scale = g*rsqrt(v+eps); bias = b - m*scale.  p = [g|b|m|v] x C
// ---------------------------------------------------------------------------
__global__ void bn_fold(const float* __restrict__ p,
                        float* __restrict__ sc, float* __restrict__ bi, int C)
{
    int c = blockIdx.x * blockDim.x + threadIdx.x;
    if (c >= C) return;
    float g = p[c], b = p[C + c], m = p[2 * C + c], v = p[3 * C + c];
    float s = g * rsqrtf(v + 1e-5f);
    sc[c] = s;
    bi[c] = b - m * s;
}

// ---------------------------------------------------------------------------
// Adaptive-avg-pool concat (k = 1,2,3,6) -> P[b,c,50]
// ---------------------------------------------------------------------------
__global__ void pool50(const float* __restrict__ X, float* __restrict__ P,
                       int Bb, int C, int H, int W)
{
    long idx = (long)blockIdx.x * blockDim.x + threadIdx.x;
    long total = (long)Bb * C * 50;
    if (idx >= total) return;
    int col = (int)(idx % 50);
    long bc = idx / 50;
    int k, cell;
    if      (col == 0) { k = 1; cell = 0; }
    else if (col < 5)  { k = 2; cell = col - 1; }
    else if (col < 14) { k = 3; cell = col - 5; }
    else               { k = 6; cell = col - 14; }
    const int bh = H / k, bw = W / k;
    const int r = cell / k, s = cell % k;
    const float* base = X + bc * (long)H * W + (long)r * bh * W + (long)s * bw;
    float sum = 0.0f;
    for (int i = 0; i < bh; ++i) {
        const float* rowp = base + (long)i * W;
        for (int j = 0; j < bw; ++j) sum += rowp[j];
    }
    P[idx] = sum / (float)(bh * bw);
}

// ---------------------------------------------------------------------------
// 50x50 linear on the token axis: out[...,j] = sum_t E[...,t]*Wl[j,t] + bl[j]
// transposeOut: 0 -> out[bc,50] ; 1 -> out[b,50,C] (row-major 50xC per batch)
// ---------------------------------------------------------------------------
__global__ void linear50(const float* __restrict__ E, const float* __restrict__ Wl,
                         const float* __restrict__ bl, float* __restrict__ out,
                         long BC, int C, int transposeOut)
{
    long idx = (long)blockIdx.x * blockDim.x + threadIdx.x;
    if (idx >= BC * 50) return;
    int  j  = (int)(idx % 50);
    long bc = idx / 50;
    const float* e = E + bc * 50;
    const float* w = Wl + (long)j * 50;
    float s = bl[j];
#pragma unroll 10
    for (int t = 0; t < 50; ++t) s += e[t] * w[t];
    if (!transposeOut) {
        out[bc * 50 + j] = s;
    } else {
        int  c = (int)(bc % C);
        long b = bc / C;
        out[((long)b * 50 + j) * C + c] = s;
    }
}

// ---------------------------------------------------------------------------
// Softmax over the 50 token rows of S[b][50][Ncols], in place (per column)
// ---------------------------------------------------------------------------
__global__ void softmax50(float* __restrict__ S, int Bb, long Ncols)
{
    long idx = (long)blockIdx.x * blockDim.x + threadIdx.x;
    if (idx >= (long)Bb * Ncols) return;
    long n = idx % Ncols;
    long b = idx / Ncols;
    float* col = S + (long)b * 50 * Ncols + n;
    float mx = -3.4e38f;
#pragma unroll 10
    for (int j = 0; j < 50; ++j) mx = fmaxf(mx, col[(long)j * Ncols]);
    float sum = 0.0f;
#pragma unroll 10
    for (int j = 0; j < 50; ++j) {
        float e = expf(col[(long)j * Ncols] - mx);
        col[(long)j * Ncols] = e;
        sum += e;
    }
    float inv = 1.0f / sum;
#pragma unroll 10
    for (int j = 0; j < 50; ++j) col[(long)j * Ncols] *= inv;
}

// ---------------------------------------------------------------------------
// Host orchestration
// ---------------------------------------------------------------------------
static inline long cdivl(long a, long b) { return (a + b - 1) / b; }

extern "C" void kernel_launch(void* const* d_in, const int* in_sizes, int n_in,
                              void* d_out, int out_size, void* d_ws, size_t ws_size,
                              hipStream_t stream)
{
    (void)in_sizes; (void)n_in; (void)out_size; (void)ws_size;
    const int  Bb = 8, C = 512, H = 72, W = 72;
    const long HW = (long)H * W;          // 5184
    const long CHW = (long)C * HW;

    const float* x        = (const float*)d_in[0];
    const float* y        = (const float*)d_in[1];
    const float* enc_x_w  = (const float*)d_in[2];   // [4,C,C]
    const float* enc_x_bn = (const float*)d_in[3];   // [4,4,C]
    const float* lin_x_w  = (const float*)d_in[4];   // [50,50]
    const float* lin_x_b  = (const float*)d_in[5];
    const float* enc_y_w  = (const float*)d_in[6];
    const float* enc_y_bn = (const float*)d_in[7];
    const float* lin_y_w  = (const float*)d_in[8];
    const float* lin_y_b  = (const float*)d_in[9];
    const float* fx_w     = (const float*)d_in[10];  // [2,C,C]
    const float* fx_bn    = (const float*)d_in[11];  // [2,4,C]
    const float* fup_w    = (const float*)d_in[12];  // [C,C]
    const float* fup_bn   = (const float*)d_in[13];  // [4,C]
    float* out = (float*)d_out;

    // workspace carve (256B aligned)
    char* wp = (char*)d_ws;
    auto alloc = [&](long nfloats) -> float* {
        float* r = (float*)wp;
        wp += ((nfloats * (long)sizeof(float) + 255) / 256) * 256;
        return r;
    };
    float* sc_encx = alloc(4 * C);   float* bi_encx = alloc(4 * C);
    float* sc_ency = alloc(4 * C);   float* bi_ency = alloc(4 * C);
    float* sc_fx   = alloc(2 * C);   float* bi_fx   = alloc(2 * C);
    float* sc_fup  = alloc(C);       float* bi_fup  = alloc(C);
    float* Px      = alloc((long)Bb * C * 50);
    float* Py      = alloc((long)Bb * C * 50);
    float* Ex      = alloc((long)Bb * C * 50);
    float* Ey      = alloc((long)Bb * C * 50);
    float* fselfL  = alloc((long)Bb * C * 50);       // [b][c][50]
    float* fyT     = alloc((long)Bb * 50 * C);       // [b][50][c]
    float* simT    = alloc((long)Bb * 50 * HW);      // [b][50][hw]
    float* big1    = alloc((long)Bb * CHW);          // fx1, later fout
    float* big2    = alloc((long)Bb * CHW);          // fx2

    // 1) fold all BN params
    for (int i = 0; i < 4; ++i)
        bn_fold<<<2, 256, 0, stream>>>(enc_x_bn + (long)i * 4 * C, sc_encx + i * C, bi_encx + i * C, C);
    for (int i = 0; i < 4; ++i)
        bn_fold<<<2, 256, 0, stream>>>(enc_y_bn + (long)i * 4 * C, sc_ency + i * C, bi_ency + i * C, C);
    for (int i = 0; i < 2; ++i)
        bn_fold<<<2, 256, 0, stream>>>(fx_bn + (long)i * 4 * C, sc_fx + i * C, bi_fx + i * C, C);
    bn_fold<<<2, 256, 0, stream>>>(fup_bn, sc_fup, bi_fup, C);

    // 2) pooled 50-token features for x and y
    {
        long tot = (long)Bb * C * 50;
        pool50<<<dim3((unsigned)cdivl(tot, 256)), 256, 0, stream>>>(x, Px, Bb, C, H, W);
        pool50<<<dim3((unsigned)cdivl(tot, 256)), 256, 0, stream>>>(y, Py, Bb, C, H, W);
    }

    // 3) encoder 1x1 convs on pooled tokens (WMMA GEMM, BN+ReLU fused)
    //    K=512, lda=512 -> fast instantiation
    const int segoff[4] = {0, 1, 5, 14};
    const int kk[4]     = {1, 2, 3, 6};
    for (int i = 0; i < 4; ++i) {
        int Ntok = kk[i] * kk[i];
        dim3 g((unsigned)cdivl(Ntok, GEMM_BN), (unsigned)cdivl(C, GEMM_BM), Bb);
        gemm_bn_wmma<false><<<g, 256, 0, stream>>>(
            enc_x_w + (long)i * C * C, C, 0,
            Px + segoff[i], 50, (long)C * 50,
            Ex + segoff[i], 50, (long)C * 50,
            sc_encx + i * C, bi_encx + i * C, nullptr, 0,
            C, Ntok, C, /*relu=*/1);
        gemm_bn_wmma<false><<<g, 256, 0, stream>>>(
            enc_y_w + (long)i * C * C, C, 0,
            Py + segoff[i], 50, (long)C * 50,
            Ey + segoff[i], 50, (long)C * 50,
            sc_ency + i * C, bi_ency + i * C, nullptr, 0,
            C, Ntok, C, /*relu=*/1);
    }

    // 4) 50x50 linears: fselfL[b,c,50], fyT[b,50,c]
    {
        long BC = (long)Bb * C;
        unsigned g = (unsigned)cdivl(BC * 50, 256);
        linear50<<<g, 256, 0, stream>>>(Ex, lin_x_w, lin_x_b, fselfL, BC, C, 0);
        linear50<<<g, 256, 0, stream>>>(Ey, lin_y_w, lin_y_b, fyT,    BC, C, 1);
    }

    // 5) fx = BN(W1 * BN(W0 * x))   (two big WMMA GEMMs, C x HW each)
    {
        dim3 g((unsigned)cdivl(HW, GEMM_BN), (unsigned)cdivl(C, GEMM_BM), Bb);
        gemm_bn_wmma<false><<<g, 256, 0, stream>>>(
            fx_w, C, 0, x, HW, CHW, big1, HW, CHW,
            sc_fx, bi_fx, nullptr, 0, C, (int)HW, C, 0);
        gemm_bn_wmma<false><<<g, 256, 0, stream>>>(
            fx_w + (long)C * C, C, 0, big1, HW, CHW, big2, HW, CHW,
            sc_fx + C, bi_fx + C, nullptr, 0, C, (int)HW, C, 0);
    }

    // 6) simT[b,50,hw] = fyT(50xC) * fx2(CxHW)   (M=50 handled by row clamp)
    {
        dim3 g((unsigned)cdivl(HW, GEMM_BN), (unsigned)cdivl(50, GEMM_BM), Bb);
        gemm_bn_wmma<false><<<g, 256, 0, stream>>>(
            fyT, C, (long)50 * C, big2, HW, CHW, simT, HW, (long)50 * HW,
            nullptr, nullptr, nullptr, 0, 50, (int)HW, C, 0);
    }

    // 7) softmax over the 50 tokens (per spatial column)
    softmax50<<<dim3((unsigned)cdivl((long)Bb * HW, 256)), 256, 0, stream>>>(simT, Bb, HW);

    // 8) foutT[b,C,hw] = fselfL(Cx50) * att(50xHW)
    //    K=50 tail + lda=50 unaligned -> EDGEK instantiation
    {
        dim3 g((unsigned)cdivl(HW, GEMM_BN), (unsigned)cdivl(C, GEMM_BM), Bb);
        gemm_bn_wmma<true><<<g, 256, 0, stream>>>(
            fselfL, 50, (long)C * 50, simT, HW, (long)50 * HW, big1, HW, CHW,
            nullptr, nullptr, nullptr, 0, C, (int)HW, 50, 0);
    }

    // 9) out = x + BN(fup_w * fout)   (residual fused in epilogue)
    {
        dim3 g((unsigned)cdivl(HW, GEMM_BN), (unsigned)cdivl(C, GEMM_BM), Bb);
        gemm_bn_wmma<false><<<g, 256, 0, stream>>>(
            fup_w, C, 0, big1, HW, CHW, out, HW, CHW,
            sc_fup, bi_fup, x, CHW, C, (int)HW, C, 0);
    }
}
